// SuP_pka_Predictor_25409026524080
// MI455X (gfx1250) — matrix operand order repaired and checked by reading, not compiled, not run on hardware
//
#include <hip/hip_runtime.h>
#include <cstddef>

// ---------------------------------------------------------------------------
// GNN (GAT + GRU) pipeline for MI455X (gfx1250, wave32, WMMA f16->f32).
// Memory-bound (~35 GFLOP vs ~2-3 GB @ 23.3 TB/s). All dense GEMMs run on
// v_wmma_f32_16x16x32_f16 with operands pre-staged in zero-K-padded f16 so
// the inner loop is pure global_load_b128 + v_wmma. B matrices staged
// transposed (Bt[N][Kpad]); he1@Wet fused with the attention scatter;
// softmax alpha pre-divided once per edge; epilogues take a wave-uniform
// full-tile fast path (no per-element EXEC juggling).
// ---------------------------------------------------------------------------

typedef __attribute__((ext_vector_type(16))) _Float16 v16h;
typedef __attribute__((ext_vector_type(8)))  _Float16 v8h;
typedef __attribute__((ext_vector_type(8)))  float    v8f;

__device__ __forceinline__ float leakyf(float x) { return x > 0.f ? x : 0.01f * x; }
__device__ __forceinline__ int   f2ord(float f) { int i = __float_as_int(f); return i < 0 ? (i ^ 0x7fffffff) : i; }
__device__ __forceinline__ float ord2f(int i)   { return __int_as_float(i < 0 ? (i ^ 0x7fffffff) : i); }
__device__ __forceinline__ float sigm(float x)  { return 1.f / (1.f + __expf(-x)); }

// Fragment loaders (contiguous k order, not the ISA table order; A and B match, wave32):
//  A 16x32 f16: lane -> row, lanes0-15 K=kt..kt+15,
//               lanes16-31 K=kt+16..kt+31
//  B 32x16 f16: lane -> col, lanes0-15 K=kt..kt+15, lanes16-31 K=kt+16..kt+31
__device__ __forceinline__ v16h load_a_frag(const _Float16* rowp, int kt, int half)
{
    const v8h* p = (const v8h*)rowp;
    v8h a0 = p[(kt >> 3) + half * 2];
    v8h a1 = p[(kt >> 3) + half * 2 + 1];
    return __builtin_shufflevector(a0, a1, 0, 1, 2, 3, 4, 5, 6, 7,
                                   8, 9, 10, 11, 12, 13, 14, 15);
}
__device__ __forceinline__ v16h load_b_frag(const _Float16* colp, int kt, int half)
{
    const v8h* p = (const v8h*)colp;
    v8h b0 = p[(kt >> 3) + half * 2];
    v8h b1 = p[(kt >> 3) + half * 2 + 1];
    return __builtin_shufflevector(b0, b1, 0, 1, 2, 3, 4, 5, 6, 7,
                                   8, 9, 10, 11, 12, 13, 14, 15);
}

// ---------------------------------------------------------------------------
// WMMA GEMM: C[M,N] = act(A16[M,KP] @ Bt16[N,KP]^T + bias[N])
// 256 threads = 8 waves; wave tile 16x16; block tile 32 rows x 64 cols.
// grid = (ceil(M/32), ceil(N/64)). KP in {64,96} -> fully unrolled.
// ---------------------------------------------------------------------------
template <int KP, int ACT, typename OutT>
__global__ void __launch_bounds__(256)
k_gemm(const _Float16* __restrict__ A, const _Float16* __restrict__ Bt,
       const float* __restrict__ bias, OutT* __restrict__ C, int M, int N)
{
    const int lane = threadIdx.x & 31;
    const int wave = threadIdx.x >> 5;
    const int half = lane >> 4;
    const int li   = lane & 15;
    const int rowBase = blockIdx.x * 32 + (wave >> 2) * 16;
    const int colBase = blockIdx.y * 64 + (wave & 3) * 16;
    if (colBase >= N) return;                 // wave-uniform

    const int arow = rowBase + li;
    const int bcol = colBase + li;
    const _Float16* ap = A  + (size_t)arow * KP;
    const _Float16* bp = Bt + (size_t)bcol * KP;

    v8f acc;
    float bv = bias[bcol];
#pragma unroll
    for (int r = 0; r < 8; ++r) acc[r] = bv;

#pragma unroll
    for (int kt = 0; kt < KP; kt += 32) {
        v16h af = load_a_frag(ap, kt, half);
        v16h bf = load_b_frag(bp, kt, half);
        acc = __builtin_amdgcn_wmma_f32_16x16x32_f16(false, af, false, bf,
                                                     (short)0, acc, false, false);
    }
    // C v8f: M = rowBase + r + 8*half, N = bcol
    OutT* cp = C + (size_t)(rowBase + half * 8) * N + bcol;
    if (rowBase + 16 <= M) {                  // wave-uniform full-tile fast path
#pragma unroll
        for (int r = 0; r < 8; ++r) {
            float v = acc[r];
            if (ACT == 1) v = leakyf(v);
            if (ACT == 2) v = fmaxf(v, 0.f);
            cp[(size_t)r * N] = (OutT)v;
        }
    } else {
#pragma unroll
        for (int r = 0; r < 8; ++r) {
            if (rowBase + half * 8 + r < M) {
                float v = acc[r];
                if (ACT == 1) v = leakyf(v);
                if (ACT == 2) v = fmaxf(v, 0.f);
                cp[(size_t)r * N] = (OutT)v;
            }
        }
    }
}

// ---------------------------------------------------------------------------
// Fused: et = he1 @ Wet + bet ; c[dst[e]] += alpha[e] * et[e]
// KP = 64, N = 64. WMMA + atomic scatter epilogue (alpha pre-divided).
// ---------------------------------------------------------------------------
__global__ void __launch_bounds__(256)
k_et_scatter(const _Float16* __restrict__ he1, const _Float16* __restrict__ BtWet,
             const float* __restrict__ bet, const float* __restrict__ alpha,
             const int* __restrict__ dst, float* __restrict__ c, int E)
{
    const int N = 64, KP = 64;
    const int lane = threadIdx.x & 31;
    const int wave = threadIdx.x >> 5;
    const int half = lane >> 4;
    const int li   = lane & 15;
    const int rowBase = blockIdx.x * 32 + (wave >> 2) * 16;
    const int colBase = (wave & 3) * 16;

    const int erow = rowBase + li;
    const int bcol = colBase + li;
    const _Float16* ap = he1   + (size_t)erow * KP;
    const _Float16* bp = BtWet + (size_t)bcol * KP;

    v8f acc;
    float bv = bet[bcol];
#pragma unroll
    for (int r = 0; r < 8; ++r) acc[r] = bv;

#pragma unroll
    for (int kt = 0; kt < KP; kt += 32) {
        v16h af = load_a_frag(ap, kt, half);
        v16h bf = load_b_frag(bp, kt, half);
        acc = __builtin_amdgcn_wmma_f32_16x16x32_f16(false, af, false, bf,
                                                     (short)0, acc, false, false);
    }
    const int rbase = rowBase + half * 8;     // this half-wave's 8 edge rows
    if (rowBase + 16 <= E) {                  // wave-uniform full-tile fast path
        // 8 contiguous dst/alpha values -> two b128 loads each
        const int4*   dp = (const int4*)(dst + rbase);
        const float4* apv = (const float4*)(alpha + rbase);
        int4   d0 = dp[0],  d1 = dp[1];
        float4 a0 = apv[0], a1 = apv[1];
        atomicAdd(&c[(size_t)d0.x * N + bcol], a0.x * acc[0]);
        atomicAdd(&c[(size_t)d0.y * N + bcol], a0.y * acc[1]);
        atomicAdd(&c[(size_t)d0.z * N + bcol], a0.z * acc[2]);
        atomicAdd(&c[(size_t)d0.w * N + bcol], a0.w * acc[3]);
        atomicAdd(&c[(size_t)d1.x * N + bcol], a1.x * acc[4]);
        atomicAdd(&c[(size_t)d1.y * N + bcol], a1.y * acc[5]);
        atomicAdd(&c[(size_t)d1.z * N + bcol], a1.z * acc[6]);
        atomicAdd(&c[(size_t)d1.w * N + bcol], a1.w * acc[7]);
    } else {
#pragma unroll
        for (int r = 0; r < 8; ++r) {
            int rr = rbase + r;
            if (rr < E) atomicAdd(&c[(size_t)dst[rr] * N + bcol], alpha[rr] * acc[r]);
        }
    }
}

// ---------------------------------------------------------------------------
// Staging conversions. grid.x = rows, blockDim.x = KP (no divisions).
// ---------------------------------------------------------------------------
__global__ void k_cvt_a(const float* __restrict__ in, _Float16* __restrict__ out, int K)
{
    int row = blockIdx.x, k = threadIdx.x, KP = blockDim.x;
    out[(size_t)row * KP + k] = (_Float16)(k < K ? in[(size_t)row * K + k] : 0.f);
}
__global__ void k_cvt_bt(const float* __restrict__ in, _Float16* __restrict__ out, int K, int N)
{
    int n = blockIdx.x, k = threadIdx.x, KP = blockDim.x;
    out[(size_t)n * KP + k] = (_Float16)(k < K ? in[(size_t)k * N + n] : 0.f);
}
// Ahe[e, 0:96] = f16(concat(nf[src[e]][0:74], ef[e][0:12], zeros[10]))
__global__ void k_build_ahe(const float* __restrict__ nf, const float* __restrict__ ef,
                            const int* __restrict__ src, _Float16* __restrict__ Ahe)
{
    int e = blockIdx.x, k = threadIdx.x;   // blockDim = 96
    float v = 0.f;
    if (k < 74)      v = nf[(size_t)src[e] * 74 + k];
    else if (k < 86) v = ef[(size_t)e * 12 + (k - 74)];
    Ahe[(size_t)e * 96 + k] = (_Float16)v;
}

// ---------------------------------------------------------------------------
// Edge attention logits + segment-max (int-punned atomicMax).
// f32 rows use 256B stride, f16 rows 128B (bases 256B-aligned) -> float4 / v8h wide loads.
// ---------------------------------------------------------------------------
__global__ void k_edge_logits(const float* rowsA, const int* __restrict__ idxA,
                              const float* rowsBf32, const _Float16* rowsBf16,
                              const int* __restrict__ idxB,
                              const float* __restrict__ W, const float* __restrict__ bias,
                              const int* __restrict__ dst,
                              float* __restrict__ lg, int* __restrict__ mord, int E)
{
    int e = blockIdx.x * blockDim.x + threadIdx.x;
    if (e >= E) return;
    const float4* ra = (const float4*)(rowsA + (size_t)idxA[e] * 64);
    const float4* w0 = (const float4*)W;
    float acc = bias[0];
#pragma unroll
    for (int i = 0; i < 16; ++i) {
        float4 a = ra[i], w = w0[i];
        acc += a.x * w.x + a.y * w.y + a.z * w.z + a.w * w.w;
    }
    if (rowsBf16) {
        const v8h* rb = (const v8h*)(rowsBf16 + (size_t)e * 64);
#pragma unroll
        for (int i = 0; i < 8; ++i) {
            v8h x = rb[i];
#pragma unroll
            for (int j = 0; j < 8; ++j) acc += (float)x[j] * W[64 + i * 8 + j];
        }
    } else {
        const float4* rb = (const float4*)(rowsBf32 + (size_t)idxB[e] * 64);
        const float4* w1 = (const float4*)(W + 64);
#pragma unroll
        for (int i = 0; i < 16; ++i) {
            float4 b = rb[i], w = w1[i];
            acc += b.x * w.x + b.y * w.y + b.z * w.z + b.w * w.w;
        }
    }
    float v = leakyf(acc);
    lg[e] = v;
    atomicMax(&mord[dst[e]], f2ord(v));
}

__global__ void k_edge_exp(const float* __restrict__ lg, const int* __restrict__ dst,
                           const int* __restrict__ mord,
                           float* __restrict__ ex, float* __restrict__ ssum, int E)
{
    int e = blockIdx.x * blockDim.x + threadIdx.x;
    if (e >= E) return;
    int d = dst[e];
    float x = __expf(lg[e] - ord2f(mord[d]));
    ex[e] = x;
    atomicAdd(&ssum[d], x);
}

// alpha[e] = ex[e] / sum[dst[e]]   (in place; one division per edge total)
__global__ void k_edge_alpha(float* __restrict__ ex, const float* __restrict__ ssum,
                             const int* __restrict__ dst, int E)
{
    int e = blockIdx.x * blockDim.x + threadIdx.x;
    if (e >= E) return;
    ex[e] = ex[e] / ssum[dst[e]];
}

// zero c, reset segment max/sum (one pass over V*64)
__global__ void k_init(float* __restrict__ c, int* __restrict__ mord,
                       float* __restrict__ ssum, int n)
{
    int t = blockIdx.x * blockDim.x + threadIdx.x;
    if (t >= n) return;
    c[t] = 0.f;
    if ((t & 63) == 0) { mord[t >> 6] = (int)0x80000000; ssum[t >> 6] = 0.f; }
}

// c[dst[e]] += alpha[e] * hp[src[e]]   (float4 per thread, 16 thr/edge)
__global__ void k_scatter_layer(const float* __restrict__ hp,
                                const float* __restrict__ alpha,
                                const int* __restrict__ src, const int* __restrict__ dst,
                                float* __restrict__ c, int E)
{
    int t = blockIdx.x * blockDim.x + threadIdx.x;
    if (t >= E * 16) return;
    int e = t >> 4, q = t & 15;
    float a = alpha[e];
    const float4 v = ((const float4*)(hp + (size_t)src[e] * 64))[q];
    float* cd = c + (size_t)dst[e] * 64 + q * 4;
    atomicAdd(cd + 0, a * v.x);
    atomicAdd(cd + 1, a * v.y);
    atomicAdd(cd + 2, a * v.z);
    atomicAdd(cd + 3, a * v.w);
}

__global__ void k_elu_inplace(float* __restrict__ c, int n)
{
    int t = blockIdx.x * blockDim.x + threadIdx.x;
    if (t >= n) return;
    float v = c[t];
    c[t] = v > 0.f ? v : (__expf(v) - 1.f);
}

// h_out = relu(gru(gi, gh, h_in)); gi/gh row stride = 192
__global__ void k_gru(const float* __restrict__ gi, const float* __restrict__ gh,
                      const float* __restrict__ hin, float* __restrict__ hout, int n)
{
    int t = blockIdx.x * blockDim.x + threadIdx.x;
    if (t >= n) return;
    int node = t >> 6, f = t & 63;
    const float* gI = gi + (size_t)node * 192;
    const float* gH = gh + (size_t)node * 192;
    float r  = sigm(gI[f]        + gH[f]);
    float z  = sigm(gI[64 + f]   + gH[64 + f]);
    float nn = tanhf(gI[128 + f] + r * gH[128 + f]);
    float hv = hin[t];
    hout[t] = fmaxf((1.f - z) * nn + z * hv, 0.f);
}

// ---------------------------------------------------------------------------
extern "C" void kernel_launch(void* const* d_in, const int* in_sizes, int n_in,
                              void* d_out, int out_size, void* d_ws, size_t ws_size,
                              hipStream_t stream)
{
    const float* node_feats = (const float*)d_in[0];
    const float* edge_feats = (const float*)d_in[1];
    const float* gc_Wpn  = (const float*)d_in[2];
    const float* gc_bpn  = (const float*)d_in[3];
    const float* gc_Wpe1 = (const float*)d_in[4];
    const float* gc_bpe1 = (const float*)d_in[5];
    const float* gc_Wpe2 = (const float*)d_in[6];
    const float* gc_bpe2 = (const float*)d_in[7];
    const float* gc_Wet  = (const float*)d_in[8];
    const float* gc_bet  = (const float*)d_in[9];
    const float* gc_Wih  = (const float*)d_in[10];
    const float* gc_Whh  = (const float*)d_in[11];
    const float* gc_bih  = (const float*)d_in[12];
    const float* gc_bhh  = (const float*)d_in[13];
    const float* gl_We   = (const float*)d_in[14];
    const float* gl_be   = (const float*)d_in[15];
    const float* gl_Wn   = (const float*)d_in[16];
    const float* gl_bn   = (const float*)d_in[17];
    const float* gl_Wih  = (const float*)d_in[18];
    const float* gl_Whh  = (const float*)d_in[19];
    const float* gl_bih  = (const float*)d_in[20];
    const float* gl_bhh  = (const float*)d_in[21];
    const int*   src     = (const int*)d_in[22];
    const int*   dst     = (const int*)d_in[23];

    const int NF = 74, G = 64, L = 2;
    const int V = in_sizes[0] / NF;
    const int E = in_sizes[22];
    const int Vpad = (V + 31) & ~31;
    const int Epad = (E + 31) & ~31;

    // ---- workspace carve ----
    char* base = (char*)d_ws;
    size_t off = 0;
    auto carve = [&](size_t bytes) -> char* {
        char* p = base + off;
        off = (off + bytes + 255) & ~(size_t)255;
        return p;
    };
    float*    hv    = (float*)carve((size_t)V * G * 4);
    float*    h     = (float*)carve((size_t)V * G * 4);
    float*    c     = (float*)carve((size_t)V * G * 4);
    float*    gi    = (float*)carve((size_t)V * 3 * G * 4);  // also hosts hp
    float*    gh    = (float*)carve((size_t)V * 3 * G * 4);
    float*    ssum  = (float*)carve((size_t)V * 4);
    int*      mord  = (int*)carve((size_t)V * 4);
    float*    lg    = (float*)carve((size_t)E * 4);
    float*    ex    = (float*)carve((size_t)E * 4);           // ex -> alpha (in place)
    _Float16* he1   = (_Float16*)carve((size_t)Epad * 64 * 2);
    _Float16* Anf   = (_Float16*)carve((size_t)Vpad * 96 * 2);
    _Float16* Ahe   = (_Float16*)carve((size_t)Epad * 96 * 2);
    _Float16* c16   = (_Float16*)carve((size_t)Vpad * 64 * 2);
    _Float16* h16   = (_Float16*)carve((size_t)Vpad * 64 * 2);
    _Float16* btPN  = (_Float16*)carve((size_t)64 * 96 * 2);
    _Float16* btPE1 = (_Float16*)carve((size_t)64 * 96 * 2);
    _Float16* btET  = (_Float16*)carve((size_t)64 * 64 * 2);
    _Float16* btIH  = (_Float16*)carve((size_t)192 * 64 * 2);
    _Float16* btHH  = (_Float16*)carve((size_t)192 * 64 * 2);
    _Float16* btWN  = (_Float16*)carve((size_t)64 * 64 * 2);
    (void)ws_size; (void)n_in; (void)out_size;

    dim3 blk(256);
    const int nVG  = V * G;
    const int bVG  = (nVG + 255) / 256;
    const int bE   = (E + 255) / 256;
    const int bE16 = (E * 16 + 255) / 256;
    dim3 gNode((V + 31) / 32, 1);              // N=64 GEMMs over nodes
    dim3 gNode3((V + 31) / 32, 3);             // N=192 GEMMs over nodes
    dim3 gEdge((E + 31) / 32, 1);              // N=64 GEMMs over edges

    // ===== staging (one-time) =====
    k_cvt_a<<<V, 96, 0, stream>>>(node_feats, Anf, NF);
    k_build_ahe<<<E, 96, 0, stream>>>(node_feats, edge_feats, src, Ahe);
    k_cvt_bt<<<64, 96, 0, stream>>>(gc_Wpn,  btPN,  NF, G);
    k_cvt_bt<<<64, 96, 0, stream>>>(gc_Wpe1, btPE1, 86, G);
    k_cvt_bt<<<64, 64, 0, stream>>>(gc_Wet,  btET,  G,  G);
    k_cvt_bt<<<192, 64, 0, stream>>>(gc_Wih, btIH,  G,  3 * G);
    k_cvt_bt<<<192, 64, 0, stream>>>(gc_Whh, btHH,  G,  3 * G);

    // ===== graph construction stage =====
    k_gemm<96, 1, float><<<gNode, blk, 0, stream>>>(Anf, btPN, gc_bpn, hv, V, G);
    k_gemm<96, 1, _Float16><<<gEdge, blk, 0, stream>>>(Ahe, btPE1, gc_bpe1, he1, E, G);
    k_init<<<bVG, blk, 0, stream>>>(c, mord, ssum, nVG);
    k_edge_logits<<<bE, blk, 0, stream>>>(hv, dst, nullptr, he1, nullptr,
                                          gc_Wpe2, gc_bpe2, dst, lg, mord, E);
    k_edge_exp<<<bE, blk, 0, stream>>>(lg, dst, mord, ex, ssum, E);
    k_edge_alpha<<<bE, blk, 0, stream>>>(ex, ssum, dst, E);
    k_et_scatter<<<gEdge, blk, 0, stream>>>(he1, btET, gc_bet, ex, dst, c, E);
    k_elu_inplace<<<bVG, blk, 0, stream>>>(c, nVG);
    k_cvt_a<<<V, 64, 0, stream>>>(c,  c16, G);
    k_cvt_a<<<V, 64, 0, stream>>>(hv, h16, G);
    k_gemm<64, 0, float><<<gNode3, blk, 0, stream>>>(c16, btIH, gc_bih, gi, V, 3 * G);
    k_gemm<64, 0, float><<<gNode3, blk, 0, stream>>>(h16, btHH, gc_bhh, gh, V, 3 * G);
    k_gru<<<bVG, blk, 0, stream>>>(gi, gh, hv, h, nVG);

    // ===== GNN layers =====
    for (int l = 0; l < L; ++l) {
        const float* We  = gl_We  + (size_t)l * 2 * G;
        const float* be  = gl_be  + l;
        const float* Wn  = gl_Wn  + (size_t)l * G * G;
        const float* bn  = gl_bn  + (size_t)l * G;
        const float* Wih = gl_Wih + (size_t)l * G * 3 * G;
        const float* Whh = gl_Whh + (size_t)l * G * 3 * G;
        const float* bih = gl_bih + (size_t)l * 3 * G;
        const float* bhh = gl_bhh + (size_t)l * 3 * G;

        k_cvt_bt<<<64, 64, 0, stream>>>(Wn,   btWN, G, G);
        k_cvt_bt<<<192, 64, 0, stream>>>(Wih, btIH, G, 3 * G);
        k_cvt_bt<<<192, 64, 0, stream>>>(Whh, btHH, G, 3 * G);

        k_init<<<bVG, blk, 0, stream>>>(c, mord, ssum, nVG);
        k_edge_logits<<<bE, blk, 0, stream>>>(h, dst, h, nullptr, src,
                                              We, be, dst, lg, mord, E);
        k_edge_exp<<<bE, blk, 0, stream>>>(lg, dst, mord, ex, ssum, E);
        k_edge_alpha<<<bE, blk, 0, stream>>>(ex, ssum, dst, E);

        k_cvt_a<<<V, 64, 0, stream>>>(h, h16, G);
        float* hp = gi;  // reuse (dead region)
        k_gemm<64, 0, float><<<gNode, blk, 0, stream>>>(h16, btWN, bn, hp, V, G);
        k_scatter_layer<<<bE16, blk, 0, stream>>>(hp, ex, src, dst, c, E);
        k_elu_inplace<<<bVG, blk, 0, stream>>>(c, nVG);
        k_cvt_a<<<V, 64, 0, stream>>>(c, c16, G);
        k_gemm<64, 0, float><<<gNode3, blk, 0, stream>>>(c16, btIH, bih, gi, V, 3 * G);
        k_gemm<64, 0, float><<<gNode3, blk, 0, stream>>>(h16, btHH, bhh, gh, V, 3 * G);
        k_gru<<<bVG, blk, 0, stream>>>(gi, gh, h, h, nVG);
    }

    hipMemcpyAsync(d_out, h, (size_t)V * G * sizeof(float),
                   hipMemcpyDeviceToDevice, stream);
}
